// DynamicGRU_24876450578772
// MI455X (gfx1250) — matrix operand order repaired and checked by reading, not compiled, not run on hardware
//
#include <hip/hip_runtime.h>

typedef __attribute__((ext_vector_type(16))) _Float16 v16h;
typedef __attribute__((ext_vector_type(8)))  _Float16 v8h;
typedef __attribute__((ext_vector_type(8)))  float    v8f;

// Sizes (fixed by the reference)
#define BATCH   32
#define SEQT    2048
#define INDIM   256
#define NNODES  256
#define ROWS    (BATCH * SEQT)          // 65536
#define GCOLS   (2 * NNODES)            // 512
#define LDSW    264                     // padded f16 row stride (16B aligned, conflict-free)

__device__ __forceinline__ v8f wmma_f16(v16h a, v16h b, v8f c) {
    return __builtin_amdgcn_wmma_f32_16x16x32_f16(
        /*neg_a=*/false, a, /*neg_b=*/false, b,
        /*c_mod=*/(short)0, c, /*reuse_a=*/false, /*reuse_b=*/false);
}

// A-operand fragment (16x32 f16) from an LDS tile with row stride `stride_h` halves.
// Per ISA 7.12.2: lanes 0-15 -> K = kt*32 + {0..7, 16..23}; lanes 16-31 -> +8.
__device__ __forceinline__ v16h load_afrag(const _Float16* base, int stride_h,
                                           int kt, int lane) {
    const int m     = lane & 15;
    const int kbase = kt * 32 + ((lane >> 4) << 3);
    const _Float16* p = base + m * stride_h + kbase;
    v8h lo = *(const v8h*)(p);
    v8h hi = *(const v8h*)(p + 16);
    v16h r;
#pragma unroll
    for (int i = 0; i < 8; ++i) { r[i] = lo[i]; r[8 + i] = hi[i]; }
    return r;
}

// ---------------------------------------------------------------------------
// Phase 0: relayout weights into B-fragment order, f32 -> f16.
// Frag element [ct][kt][lane][e] = W[kt*32 + (e>>3)*16 + (lane>>4)*8 + (e&7)][ct*16 + (lane&15)]
// ---------------------------------------------------------------------------
__global__ __launch_bounds__(256) void gru_convert_w(
    const float* __restrict__ W_g, const float* __restrict__ W_c,
    _Float16* __restrict__ Wgx_f, _Float16* __restrict__ Wgh_f,
    _Float16* __restrict__ Wcx_f, _Float16* __restrict__ Wch_f)
{
    int id = blockIdx.x * 256 + threadIdx.x;        // 0 .. 393215
    if (id < 262144) {                              // W_g: 2 x 131072 frag elements
        int local = id & 131071;
        int e  = local & 15;
        int ln = (local >> 4) & 31;
        int kt = (local >> 9) & 7;
        int ct = local >> 12;                       // 0..31
        int k  = kt * 32 + ((e >> 3) << 4) + ((ln >> 4) << 3) + (e & 7);    // 0..255
        int n  = ct * 16 + (ln & 15);               // 0..511
        int rowoff = (id < 131072) ? 0 : INDIM;
        _Float16 v = (_Float16)W_g[(size_t)(k + rowoff) * GCOLS + n];
        ((id < 131072) ? Wgx_f : Wgh_f)[local] = v;
    } else {                                        // W_c: 2 x 65536 frag elements
        int id2   = id - 262144;
        int local = id2 & 65535;
        int e  = local & 15;
        int ln = (local >> 4) & 31;
        int kt = (local >> 9) & 7;
        int ct = local >> 12;                       // 0..15
        int k  = kt * 32 + ((e >> 3) << 4) + ((ln >> 4) << 3) + (e & 7);
        int n  = ct * 16 + (ln & 15);               // 0..255
        int rowoff = (id2 < 65536) ? 0 : INDIM;
        _Float16 v = (_Float16)W_c[(size_t)(k + rowoff) * NNODES + n];
        ((id2 < 65536) ? Wcx_f : Wch_f)[local] = v;
    }
}

// ---------------------------------------------------------------------------
// Phase 1: Gx = X @ Wgx + b_g  (65536x512), Cx = X @ Wcx + b_c (65536x256)
// One block per 16-row strip; 8 waves x 6 column tiles; K=256 (8 wmma each).
// ---------------------------------------------------------------------------
__global__ __launch_bounds__(256) void gru_phase1(
    const float* __restrict__ X,
    const float* __restrict__ b_g, const float* __restrict__ b_c,
    const _Float16* __restrict__ Wgx_f, const _Float16* __restrict__ Wcx_f,
    float* __restrict__ Gx, float* __restrict__ Cx)
{
    __shared__ __align__(16) _Float16 Xs[16 * LDSW];
    const int tid = threadIdx.x;
    const int strip = blockIdx.x;                   // 0..4095

    {   // stage 16x256 of X into LDS as f16
        int r  = tid >> 4;
        int c0 = (tid & 15) * 16;
        const float* src = X + ((size_t)strip * 16 + r) * INDIM + c0;
#pragma unroll
        for (int i = 0; i < 16; ++i) Xs[r * LDSW + c0 + i] = (_Float16)src[i];
    }
    __syncthreads();

    const int lane = tid & 31, wave = tid >> 5;
    v16h afrag[8];
#pragma unroll
    for (int kt = 0; kt < 8; ++kt) afrag[kt] = load_afrag(Xs, LDSW, kt, lane);

    const int mrow = (lane >> 4) << 3;
    const int n15  = lane & 15;

#pragma unroll
    for (int i = 0; i < 6; ++i) {
        int ct = wave + i * 8;                      // 0..47 (0-31: Gx, 32-47: Cx)
        const _Float16* bf = (ct < 32) ? (Wgx_f + (size_t)ct * 8 * 512)
                                       : (Wcx_f + (size_t)(ct - 32) * 8 * 512);
        v8f acc = {};
#pragma unroll
        for (int kt = 0; kt < 8; ++kt) {
            v16h b = *(const v16h*)(bf + ((size_t)kt * 32 + lane) * 16);
            acc = wmma_f16(afrag[kt], b, acc);
        }
        size_t row0 = (size_t)strip * 16 + mrow;
        if (ct < 32) {
            int col = ct * 16 + n15;
            float bias = b_g[col];
#pragma unroll
            for (int e = 0; e < 8; ++e) Gx[(row0 + e) * GCOLS + col] = acc[e] + bias;
        } else {
            int col = (ct - 32) * 16 + n15;
            float bias = b_c[col];
#pragma unroll
            for (int e = 0; e < 8; ++e) Cx[(row0 + e) * NNODES + col] = acc[e] + bias;
        }
    }
}

// ---------------------------------------------------------------------------
// Phase 2: serial scan. 2 blocks, each owns 16 batch rows; h lives in LDS.
// Per step: gates = sigmoid(h@Wgh + Gx_t); c = tanh((r*h)@Wch + Cx_t);
//           h = u*h + (1-u)*c; write h to out.
// ---------------------------------------------------------------------------
__global__ __launch_bounds__(256) void gru_phase2(
    const float* __restrict__ h0,
    const _Float16* __restrict__ Wgh_f, const _Float16* __restrict__ Wch_f,
    const float* __restrict__ Gx, const float* __restrict__ Cx,
    float* __restrict__ out)
{
    __shared__ float h32[16 * NNODES];
    __shared__ float uu [16 * NNODES];
    __shared__ __align__(16) _Float16 h16 [16 * LDSW];
    __shared__ __align__(16) _Float16 rh16[16 * LDSW];

    const int tid = threadIdx.x, lane = tid & 31, wave = tid >> 5;
    const int b0 = blockIdx.x * 16;
    const int mrow = (lane >> 4) << 3;
    const int n15  = lane & 15;

    for (int i = tid; i < 16 * NNODES; i += 256) {
        int m = i >> 8, n = i & 255;
        float v = h0[(size_t)(b0 + m) * NNODES + n];
        h32[i] = v;
        h16[m * LDSW + n] = (_Float16)v;
    }
    __syncthreads();

#pragma unroll 1
    for (int t = 0; t < SEQT; ++t) {
        // ---- prefetch next step's Gx/Cx rows (global_prefetch_b8) ----
        if (t + 1 < SEQT) {
            int r = tid >> 4, s = tid & 15;         // 16 rows x 16 x 128B = 32KB
            __builtin_prefetch(Gx + ((size_t)(b0 + r) * SEQT + (t + 1)) * GCOLS + s * 32, 0, 0);
            if (tid < 128) {
                int r2 = tid >> 3, s2 = tid & 7;    // 16 rows x 8 x 128B = 16KB
                __builtin_prefetch(Cx + ((size_t)(b0 + r2) * SEQT + (t + 1)) * NNODES + s2 * 32, 0, 0);
            }
        }

        // ---- gates: [16,256] @ [256,512] ; 4 column tiles per wave ----
        v16h afr[8];
#pragma unroll
        for (int kt = 0; kt < 8; ++kt) afr[kt] = load_afrag(h16, LDSW, kt, lane);

#pragma unroll
        for (int i = 0; i < 4; ++i) {
            int ct  = wave * 4 + i;                 // 0..31
            int col = ct * 16 + n15;
            // issue Gx loads before the wmma chain so they overlap it
            float gxv[8];
            size_t growb = (size_t)(b0 + mrow) * SEQT + t;
#pragma unroll
            for (int e = 0; e < 8; ++e) gxv[e] = Gx[(growb + (size_t)e * SEQT) * GCOLS + col];

            const _Float16* bf = Wgh_f + (size_t)ct * 8 * 512;
            v8f acc = {};
#pragma unroll
            for (int kt = 0; kt < 8; ++kt) {
                v16h b = *(const v16h*)(bf + ((size_t)kt * 32 + lane) * 16);
                acc = wmma_f16(afr[kt], b, acc);
            }
#pragma unroll
            for (int e = 0; e < 8; ++e) {
                int m = mrow + e;
                float g = 1.0f / (1.0f + __expf(-(acc[e] + gxv[e])));
                if (ct < 16) {                      // r gate -> r*h (f16, A-operand)
                    rh16[m * LDSW + col] = (_Float16)(g * h32[m * NNODES + col]);
                } else {                            // u gate
                    uu[m * NNODES + (col - NNODES)] = g;
                }
            }
        }
        __syncthreads();

        // ---- candidate + state update: [16,256] @ [256,256]; 2 tiles/wave ----
#pragma unroll
        for (int kt = 0; kt < 8; ++kt) afr[kt] = load_afrag(rh16, LDSW, kt, lane);

#pragma unroll
        for (int i = 0; i < 2; ++i) {
            int ct  = wave * 2 + i;                 // 0..15
            int col = ct * 16 + n15;
            float cxv[8];
            size_t growb = (size_t)(b0 + mrow) * SEQT + t;
#pragma unroll
            for (int e = 0; e < 8; ++e) cxv[e] = Cx[(growb + (size_t)e * SEQT) * NNODES + col];

            const _Float16* bf = Wch_f + (size_t)ct * 8 * 512;
            v8f acc = {};
#pragma unroll
            for (int kt = 0; kt < 8; ++kt) {
                v16h b = *(const v16h*)(bf + ((size_t)kt * 32 + lane) * 16);
                acc = wmma_f16(afr[kt], b, acc);
            }
#pragma unroll
            for (int e = 0; e < 8; ++e) {
                int m = mrow + e;
                size_t grow = growb + (size_t)e * SEQT;
                float c  = tanhf(acc[e] + cxv[e]);
                float u  = uu[m * NNODES + col];
                float hp = h32[m * NNODES + col];
                float hn = u * hp + (1.0f - u) * c;
                out[grow * NNODES + col] = hn;
                h32[m * NNODES + col] = hn;
                h16[m * LDSW + col]   = (_Float16)hn;
            }
        }
        __syncthreads();                            // h ready for next step
    }
}

// ---------------------------------------------------------------------------
extern "C" void kernel_launch(void* const* d_in, const int* in_sizes, int n_in,
                              void* d_out, int out_size, void* d_ws, size_t ws_size,
                              hipStream_t stream) {
    const float* x   = (const float*)d_in[0];   // [65536, 256]
    const float* h0  = (const float*)d_in[1];   // [32, 256]
    const float* W_g = (const float*)d_in[2];   // [512, 512]
    const float* b_g = (const float*)d_in[3];   // [512]
    const float* W_c = (const float*)d_in[4];   // [512, 256]
    const float* b_c = (const float*)d_in[5];   // [256]
    float* out = (float*)d_out;                 // [65536, 256]

    // workspace layout
    float*    Gx    = (float*)d_ws;                                  // 65536*512 f32
    float*    Cx    = Gx + (size_t)ROWS * GCOLS;                     // 65536*256 f32
    _Float16* Wgx_f = (_Float16*)(Cx + (size_t)ROWS * NNODES);       // 131072 f16
    _Float16* Wgh_f = Wgx_f + 131072;
    _Float16* Wcx_f = Wgh_f + 131072;
    _Float16* Wch_f = Wcx_f + 65536;

    gru_convert_w<<<dim3(1536), dim3(256), 0, stream>>>(W_g, W_c,
                                                        Wgx_f, Wgh_f, Wcx_f, Wch_f);
    gru_phase1<<<dim3(ROWS / 16), dim3(256), 0, stream>>>(x, b_g, b_c,
                                                          Wgx_f, Wcx_f, Gx, Cx);
    gru_phase2<<<dim3(BATCH / 16), dim3(256), 0, stream>>>(h0, Wgh_f, Wch_f,
                                                           Gx, Cx, out);
}
